// HeteroNodeEncoder_19731079758358
// MI455X (gfx1250) — matrix operand order repaired and checked by reading, not compiled, not run on hardware
//
#include <hip/hip_runtime.h>

// GraphSAGE 2-layer forward for MI455X (gfx1250, wave32).
// - Node features kept in bf16 (halves random-gather traffic on 1.6M edges;
//   25.6MB feature matrix stays L2-resident for the gather).
// - GEMMs: v_wmma_f32_16x16x32_bf16, A/B fragments loaded directly from
//   row-major bf16 global memory (no LDS staging, no cvt in the hot loop).
// - Edge aggregation: one wave32 per edge, f32 atomics into L2-resident agg.
// - agg/deg zeroing fused into the GEMM kernels that precede each scatter.
// - Uniform fast-path epilogues: 8 stores with immediate offsets, no guards.

typedef __attribute__((ext_vector_type(16))) __bf16 v16bf;
typedef __attribute__((ext_vector_type(8)))  __bf16 v8bf;
typedef __attribute__((ext_vector_type(4)))  __bf16 v4bf;
typedef __attribute__((ext_vector_type(8)))  float  v8f;

#define D 128
#define TILE_M 16

// ---------------------------------------------------------------------------
// WMMA fragment loaders (CDNA5 16-bit layouts, CDNA5 ISA):
//  A (16x32): lane holds M=lane&15; K in [kk+half*8, +8) and [kk+16+half*8, +8)
//  B (32x16): lane holds N=lane&15; K in [kk+half*8, +8) and [kk+16+half*8, +8)
//  C/D (16x16 f32): lane holds col N=lane&15, rows half*8 + j
// Works for both global and LDS bf16 pointers (addrspace inferred).
// ---------------------------------------------------------------------------
__device__ __forceinline__ v16bf load_a_frag(const __bf16* row, int half, int kk) {
  v8bf lo = *(const v8bf*)(row + kk + half * 8);
  v8bf hi = *(const v8bf*)(row + kk + 16 + half * 8);
  return __builtin_shufflevector(lo, hi, 0, 1, 2, 3, 4, 5, 6, 7,
                                 8, 9, 10, 11, 12, 13, 14, 15);
}

__device__ __forceinline__ v16bf load_b_frag(const __bf16* row, int half, int kk) {
  const __bf16* p = row + kk + half * 8;
  v8bf lo = *(const v8bf*)(p);
  v8bf hi = *(const v8bf*)(p + 16);
  return __builtin_shufflevector(lo, hi, 0, 1, 2, 3, 4, 5, 6, 7,
                                 8, 9, 10, 11, 12, 13, 14, 15);
}

// ---------------------------------------------------------------------------
// f32 -> bf16 conversion for x (one-time per launch)
// ---------------------------------------------------------------------------
__global__ __launch_bounds__(256) void cvt_f32_bf16_kernel(
    const float* __restrict__ src, __bf16* __restrict__ dst, size_t n) {
  size_t i = ((size_t)blockIdx.x * blockDim.x + threadIdx.x) * 4;
  size_t stride = (size_t)gridDim.x * blockDim.x * 4;
  for (; i + 3 < n; i += stride) {
    float4 v = *(const float4*)(src + i);
    v4bf o;
    o[0] = (__bf16)v.x; o[1] = (__bf16)v.y; o[2] = (__bf16)v.z; o[3] = (__bf16)v.w;
    *(v4bf*)(dst + i) = o;
  }
}

// All six 128x128 weight matrices -> bf16, one launch (6*16384/1024 = 96 blocks)
__global__ __launch_bounds__(256) void cvt_weights_kernel(
    const float* __restrict__ s0, const float* __restrict__ s1,
    const float* __restrict__ s2, const float* __restrict__ s3,
    const float* __restrict__ s4, const float* __restrict__ s5,
    __bf16* __restrict__ dst) {
  int idx = (blockIdx.x * 256 + threadIdx.x) * 4;
  int m = idx >> 14;            // which matrix (16384 f32 each)
  int off = idx & 16383;
  const float* s = (m == 0) ? s0 : (m == 1) ? s1 : (m == 2) ? s2
                 : (m == 3) ? s3 : (m == 4) ? s4 : s5;
  float4 v = *(const float4*)(s + off);
  v4bf o;
  o[0] = (__bf16)v.x; o[1] = (__bf16)v.y; o[2] = (__bf16)v.z; o[3] = (__bf16)v.w;
  *(v4bf*)(dst + idx) = o;
}

// ---------------------------------------------------------------------------
// Y = relu(X @ W^T + b)  (bf16 in, bf16 out, f32 accumulate)
// Also zeroes the agg/deg rows this block owns, for the scatter that follows.
// Block = 256 threads = 8 waves; 16 rows x 128 cols per block.
// ---------------------------------------------------------------------------
__global__ __launch_bounds__(256) void gemm_bias_relu_kernel(
    const __bf16* __restrict__ X, const __bf16* __restrict__ W,
    const float* __restrict__ bias, __bf16* __restrict__ Y,
    float* __restrict__ agg_zero, float* __restrict__ deg_zero, int nrows) {
  const int tid = threadIdx.x;
  const int rowbase = blockIdx.x * TILE_M;
  const bool full = (rowbase + TILE_M <= nrows);   // uniform across block

  // Fused zeroing of agg[rowbase..rowbase+16) and deg (16*128 f32 = 512 float4)
  if (full) {
    float4 z = {0.0f, 0.0f, 0.0f, 0.0f};
    float4* az = (float4*)(agg_zero + (size_t)rowbase * D);
    az[tid] = z;
    az[tid + 256] = z;
    if (tid < TILE_M) deg_zero[rowbase + tid] = 0.0f;
  } else {
    for (int i = tid; i < TILE_M * D; i += 256) {
      int gr = rowbase + (i >> 7);
      if (gr < nrows) agg_zero[(size_t)gr * D + (i & 127)] = 0.0f;
    }
    if (tid < TILE_M && rowbase + tid < nrows) deg_zero[rowbase + tid] = 0.0f;
  }

  const int wave = tid >> 5;
  const int lane = tid & 31;
  const int half = lane >> 4;
  const int mn = lane & 15;
  const int colbase = wave * 16;

  int arow = rowbase + mn;
  if (arow >= nrows) arow = nrows - 1;             // clamp (keeps EXEC all-ones)
  const __bf16* Xrow = X + (size_t)arow * D;
  const __bf16* Wrow = W + (size_t)(colbase + mn) * D;

  v8f acc = {};
#pragma unroll
  for (int kk = 0; kk < D; kk += 32) {
    v16bf a = load_a_frag(Xrow, half, kk);
    v16bf b = load_b_frag(Wrow, half, kk);
    acc = __builtin_amdgcn_wmma_f32_16x16x32_bf16(false, a, false, b,
                                                  (short)0, acc, false, false);
  }

  const float bv = bias[colbase + mn];
  if (full) {
    __bf16* base = Y + (size_t)(rowbase + half * 8) * D + colbase + mn;
#pragma unroll
    for (int j = 0; j < 8; ++j) {
      float v = acc[j] + bv;
      base[(size_t)j * D] = (__bf16)(v > 0.0f ? v : 0.0f);
    }
  } else {
#pragma unroll
    for (int j = 0; j < 8; ++j) {
      int gr = rowbase + half * 8 + j;
      if (gr < nrows) {
        float v = acc[j] + bv;
        Y[(size_t)gr * D + colbase + mn] = (__bf16)(v > 0.0f ? v : 0.0f);
      }
    }
  }
}

// ---------------------------------------------------------------------------
// Y = relu( (agg/deg) @ Wl^T + bl + H @ Wr^T )  (fused SAGEConv combine)
// H fragments direct from global bf16; mean tile staged via LDS (f32 agg).
// OutT = __bf16 for conv1 (feeds next layer) or float for final output.
// ---------------------------------------------------------------------------
template <typename OutT>
__global__ __launch_bounds__(256) void sage_combine_kernel(
    const __bf16* __restrict__ H, const float* __restrict__ Agg,
    const float* __restrict__ Deg,
    const __bf16* __restrict__ Wl, const float* __restrict__ bl,
    const __bf16* __restrict__ Wr, OutT* __restrict__ Y, int nrows) {
  __shared__ __bf16 Ms[TILE_M * D];
  __shared__ float Dinv[TILE_M];
  const int tid = threadIdx.x;
  const int rowbase = blockIdx.x * TILE_M;
  const bool full = (rowbase + TILE_M <= nrows);   // uniform across block

  if (tid < TILE_M) {
    int gr = rowbase + tid;
    float dg = (gr < nrows) ? Deg[gr] : 1.0f;
    Dinv[tid] = 1.0f / fmaxf(dg, 1.0f);
  }
  __syncthreads();

  if (full) {
    // 16*128 = 2048 elems, 8 per thread, contiguous float4 reads of Agg
    const float4* ag = (const float4*)(Agg + (size_t)rowbase * D);
    v4bf* ms = (v4bf*)Ms;
#pragma unroll
    for (int t = 0; t < 2; ++t) {
      int i = tid + t * 256;               // float4 index within tile
      float4 a = ag[i];
      float di = Dinv[i >> 5];             // 32 float4 per row
      v4bf o;
      o[0] = (__bf16)(a.x * di); o[1] = (__bf16)(a.y * di);
      o[2] = (__bf16)(a.z * di); o[3] = (__bf16)(a.w * di);
      ms[i] = o;
    }
  } else {
    for (int i = tid; i < TILE_M * D; i += 256) {
      int r = i >> 7, k = i & 127;
      int gr = rowbase + r;
      float mv = (gr < nrows) ? Agg[(size_t)gr * D + k] * Dinv[r] : 0.0f;
      Ms[i] = (__bf16)mv;
    }
  }
  __syncthreads();

  const int wave = tid >> 5;
  const int lane = tid & 31;
  const int half = lane >> 4;
  const int mn = lane & 15;
  const int colbase = wave * 16;

  int arow = rowbase + mn;
  if (arow >= nrows) arow = nrows - 1;
  const __bf16* Hrow = H + (size_t)arow * D;
  const __bf16* Mrow = Ms + mn * D;
  const __bf16* WlRow = Wl + (size_t)(colbase + mn) * D;
  const __bf16* WrRow = Wr + (size_t)(colbase + mn) * D;

  v8f acc = {};
#pragma unroll
  for (int kk = 0; kk < D; kk += 32) {
    v16bf ah = load_a_frag(Hrow, half, kk);
    v16bf br = load_b_frag(WrRow, half, kk);
    acc = __builtin_amdgcn_wmma_f32_16x16x32_bf16(false, ah, false, br,
                                                  (short)0, acc, false, false);
    v16bf am = load_a_frag(Mrow, half, kk);
    v16bf blf = load_b_frag(WlRow, half, kk);
    acc = __builtin_amdgcn_wmma_f32_16x16x32_bf16(false, am, false, blf,
                                                  (short)0, acc, false, false);
  }

  const float bv = bl[colbase + mn];
  if (full) {
    OutT* base = Y + (size_t)(rowbase + half * 8) * D + colbase + mn;
#pragma unroll
    for (int j = 0; j < 8; ++j) {
      float v = acc[j] + bv;
      base[(size_t)j * D] = (OutT)(v > 0.0f ? v : 0.0f);
    }
  } else {
#pragma unroll
    for (int j = 0; j < 8; ++j) {
      int gr = rowbase + half * 8 + j;
      if (gr < nrows) {
        float v = acc[j] + bv;
        Y[(size_t)gr * D + colbase + mn] = (OutT)(v > 0.0f ? v : 0.0f);
      }
    }
  }
}

// ---------------------------------------------------------------------------
// One wave per edge: gather 128-bf16 row of h[src] (8B/lane), f32 atomic-add
// into agg[dst] (resolves in L2 — agg is 51MB < 192MB L2). Lane0 bumps deg.
// ---------------------------------------------------------------------------
__global__ __launch_bounds__(256) void scatter_sum_kernel(
    const __bf16* __restrict__ H, const long long* __restrict__ ei,
    float* __restrict__ agg, float* __restrict__ deg, int nedges) {
  int gwave = (int)(((size_t)blockIdx.x * 256 + threadIdx.x) >> 5);
  int lane = threadIdx.x & 31;
  if (gwave >= nedges) return;
  long long s = ei[gwave];
  long long d = ei[(size_t)nedges + gwave];
  v4bf v = ((const v4bf*)(H + (size_t)s * D))[lane];
  float* ap = agg + (size_t)d * D + lane * 4;
  unsafeAtomicAdd(ap + 0, (float)v[0]);
  unsafeAtomicAdd(ap + 1, (float)v[1]);
  unsafeAtomicAdd(ap + 2, (float)v[2]);
  unsafeAtomicAdd(ap + 3, (float)v[3]);
  if (lane == 0) unsafeAtomicAdd(deg + d, 1.0f);
}

// ---------------------------------------------------------------------------
extern "C" void kernel_launch(void* const* d_in, const int* in_sizes, int n_in,
                              void* d_out, int out_size, void* d_ws, size_t ws_size,
                              hipStream_t stream) {
  const float* x      = (const float*)d_in[0];
  const long long* ei = (const long long*)d_in[1];  // int64 edge_index [2,E]
  const float* W1     = (const float*)d_in[2];
  const float* b1     = (const float*)d_in[3];
  const float* W2     = (const float*)d_in[4];
  const float* b2     = (const float*)d_in[5];
  const float* c1_Wl  = (const float*)d_in[6];
  const float* c1_bl  = (const float*)d_in[7];
  const float* c1_Wr  = (const float*)d_in[8];
  const float* c2_Wl  = (const float*)d_in[9];
  const float* c2_bl  = (const float*)d_in[10];
  const float* c2_Wr  = (const float*)d_in[11];

  const int N = in_sizes[0] / D;
  const int E = in_sizes[1] / 2;
  const size_t ND = (size_t)N * D;
  const size_t WSZ = (size_t)D * D;   // 16384

  // Workspace: xb | hA | hB (bf16 features), 6x bf16 weights, agg | deg (f32)
  __bf16* xb = (__bf16*)d_ws;
  __bf16* hA = xb + ND;
  __bf16* hB = hA + ND;
  __bf16* wq = hB + ND;               // 6 weight matrices, bf16, contiguous
  __bf16* w1b   = wq + 0 * WSZ;
  __bf16* w2b   = wq + 1 * WSZ;
  __bf16* c1Wlb = wq + 2 * WSZ;
  __bf16* c1Wrb = wq + 3 * WSZ;
  __bf16* c2Wlb = wq + 4 * WSZ;
  __bf16* c2Wrb = wq + 5 * WSZ;
  float* agg = (float*)(wq + 6 * WSZ);
  float* deg = agg + ND;

  const int gemmBlocks = (N + TILE_M - 1) / TILE_M;   // 6250
  const int scatBlocks = (E + 7) / 8;                 // 8 edges (waves) per block
  float* out = (float*)d_out;

  // One-time conversions to bf16 (2 launches)
  cvt_f32_bf16_kernel<<<2048, 256, 0, stream>>>(x, xb, ND);
  cvt_weights_kernel<<<96, 256, 0, stream>>>(W1, W2, c1_Wl, c1_Wr, c2_Wl, c2_Wr, wq);

  // h = relu(x @ W1^T + b1)           (also zeroes agg/deg for scatter #1)
  gemm_bias_relu_kernel<<<gemmBlocks, 256, 0, stream>>>(xb, w1b, b1, hA,
                                                        agg, deg, N);
  // SAGEConv1
  scatter_sum_kernel<<<scatBlocks, 256, 0, stream>>>(hA, ei, agg, deg, E);
  sage_combine_kernel<__bf16><<<gemmBlocks, 256, 0, stream>>>(
      hA, agg, deg, c1Wlb, c1_bl, c1Wrb, hB, N);

  // h = relu(h @ W2^T + b2)           (also re-zeroes agg/deg for scatter #2)
  gemm_bias_relu_kernel<<<gemmBlocks, 256, 0, stream>>>(hB, w2b, b2, hA,
                                                        agg, deg, N);
  // SAGEConv2 -> d_out (f32)
  scatter_sum_kernel<<<scatBlocks, 256, 0, stream>>>(hA, ei, agg, deg, E);
  sage_combine_kernel<float><<<gemmBlocks, 256, 0, stream>>>(
      hA, agg, deg, c2Wlb, c2_bl, c2Wrb, out, N);
}
